// EncoderLayer_27745488732663
// MI455X (gfx1250) — compile-verified, matrix operand order checked by reading, not run on hardware
//
#include <hip/hip_runtime.h>
#include <math.h>

#define RES 12000
#define KNB 48
#define HD  128
#define LDX 392   // 384 + 8 (bf16 elements)
#define LDA 136   // 128 + 8
#define LDF 520   // 512 + 8
#define LDD 132   // 128 + 4 (f32)
#define EPS 1e-5f

typedef __attribute__((ext_vector_type(16))) __bf16 v16bf;
typedef __attribute__((ext_vector_type(8)))  __bf16 v8bf;
typedef __attribute__((ext_vector_type(8)))  float  v8f;

__device__ __forceinline__ float gelu_exact(float x) {
    return 0.5f * x * (1.0f + erff(x * 0.70710678118654752f));
}

// Load a 16x32 bf16 A/B fragment from a row-major [rows][ld] matrix.
// lane 0-15:  row = rbase+lane,    elems 0-7 = K kbase+0..7,   8-15 = K kbase+16..23
// lane 16-31: row = rbase+lane-16, elems 0-7 = K kbase+8..15,  8-15 = K kbase+24..31
__device__ __forceinline__ v16bf frag_ld(const __bf16* M, int ld, int rbase, int kbase, int lane) {
    const int r  = rbase + (lane & 15);
    const int ko = kbase + ((lane & 16) ? 8 : 0);
    const __bf16* p = M + r * ld + ko;
    v8bf lo = *(const v8bf*)(p);
    v8bf hi = *(const v8bf*)(p + 16);
    return __builtin_shufflevector(lo, hi, 0,1,2,3,4,5,6,7,8,9,10,11,12,13,14,15);
}

__device__ __forceinline__ v8f wmma_bf16(v16bf a, v16bf b, v8f c) {
    return __builtin_amdgcn_wmma_f32_16x16x32_bf16(false, a, false, b, (short)0, c, false, false);
}

// C tile store: lane L -> col = colbase + (L&15), reg q -> row = q + (L&16 ? 8 : 0)
template<bool GELU>
__device__ __forceinline__ void store_act_bf(const v8f& c0, const v8f& c1, const v8f& c2,
        const float* bias, __bf16* out, int ld, int colbase, int lane) {
    const int col  = colbase + (lane & 15);
    const int roff = (lane & 16) ? 8 : 0;
    const float bv = bias[col];
    #pragma unroll
    for (int q = 0; q < 8; ++q) {
        float v0 = c0[q] + bv, v1 = c1[q] + bv, v2 = c2[q] + bv;
        if (GELU) { v0 = gelu_exact(v0); v1 = gelu_exact(v1); v2 = gelu_exact(v2); }
        out[(q + roff) * ld + col]      = (__bf16)v0;
        out[(16 + q + roff) * ld + col] = (__bf16)v1;
        out[(32 + q + roff) * ld + col] = (__bf16)v2;
    }
}

__device__ __forceinline__ void store_f32_ld(const v8f& c0, const v8f& c1, const v8f& c2,
        const float* bias, float* out, int ld, int colbase, int lane) {
    const int col  = colbase + (lane & 15);
    const int roff = (lane & 16) ? 8 : 0;
    const float bv = bias[col];
    #pragma unroll
    for (int q = 0; q < 8; ++q) {
        out[(q + roff) * ld + col]      = c0[q] + bv;
        out[(16 + q + roff) * ld + col] = c1[q] + bv;
        out[(32 + q + roff) * ld + col] = c2[q] + bv;
    }
}

// [48, KD] (LDS, bf16) @ WT[N=128 rows][KD] (global bf16) -> 3 M-tiles per wave's N strip
template<int KD>
__device__ __forceinline__ void gemm48(const __bf16* Xin, int ldx, const __bf16* WT,
                                       int wv, int lane, v8f& c0, v8f& c1, v8f& c2) {
    const int n0 = wv * 16;
    #pragma unroll
    for (int kt = 0; kt < KD; kt += 32) {
        v16bf bfr = frag_ld(WT, KD, n0, kt, lane);
        v16bf a0  = frag_ld(Xin, ldx, 0,  kt, lane);
        v16bf a1  = frag_ld(Xin, ldx, 16, kt, lane);
        v16bf a2  = frag_ld(Xin, ldx, 32, kt, lane);
        c0 = wmma_bf16(a0, bfr, c0);
        c1 = wmma_bf16(a1, bfr, c1);
        c2 = wmma_bf16(a2, bfr, c2);
    }
}

__device__ __forceinline__ float wave_sum(float v) {
    #pragma unroll
    for (int m = 16; m >= 1; m >>= 1) v += __shfl_xor(v, m, 32);
    return v;
}

// ---------------- prep kernels ----------------
__global__ void convert_bf16_kernel(const float* __restrict__ src, __bf16* __restrict__ dst, int n) {
    int idx = blockIdx.x * 256 + threadIdx.x;
    if (idx < n) dst[idx] = (__bf16)src[idx];
}

// src [rows][cols] f32 -> dst [cols][rows] bf16
__global__ void transpose_bf16_kernel(const float* __restrict__ src, __bf16* __restrict__ dst,
                                      int rows, int cols) {
    int idx = blockIdx.x * 256 + threadIdx.x;
    if (idx < rows * cols) {
        int r = idx / cols, c = idx % cols;
        dst[c * rows + r] = (__bf16)src[r * cols + c];
    }
}

// ---------------- phase 1: message MLP + sum + LN1 ----------------
__global__ __launch_bounds__(256) void msg_node_kernel(
    const float* __restrict__ node_h, const float* __restrict__ edge_h,
    const int* __restrict__ edge_idx, const __bf16* __restrict__ nodeb,
    const __bf16* __restrict__ w0T, const __bf16* __restrict__ w1T, const __bf16* __restrict__ w2T,
    const float* __restrict__ b0, const float* __restrict__ b1, const float* __restrict__ b2,
    const float* __restrict__ ln1_g, const float* __restrict__ ln1_b,
    float* __restrict__ node1f, __bf16* __restrict__ node1b)
{
    __shared__ __attribute__((aligned(16))) __bf16 X[KNB * LDX];
    __shared__ __attribute__((aligned(16))) __bf16 act[KNB * LDA];
    __shared__ __attribute__((aligned(16))) float red[256];
    __shared__ __attribute__((aligned(16))) float nodev[HD];

    const int i = blockIdx.x;
    const int t = threadIdx.x;
    const int lane = t & 31, wv = t >> 5;

    // X = [h_i | h_j | e_ij]  (48 x 384 bf16)
    for (int e = t; e < KNB * HD; e += 256) {
        const int r = e >> 7, c = e & 127;
        const int j = edge_idx[i * KNB + r];
        X[r * LDX + c]          = nodeb[i * HD + c];
        X[r * LDX + HD + c]     = nodeb[j * HD + c];
        X[r * LDX + 2 * HD + c] = (__bf16)edge_h[((size_t)i * KNB + r) * HD + c];
    }
    __syncthreads();

    v8f c0 = {}, c1 = {}, c2 = {};
    gemm48<384>(X, LDX, w0T, wv, lane, c0, c1, c2);
    store_act_bf<true>(c0, c1, c2, b0, act, LDA, wv * 16, lane);
    __syncthreads();

    v8f d0 = {}, d1 = {}, d2 = {};
    gemm48<128>(act, LDA, w1T, wv, lane, d0, d1, d2);
    __syncthreads();
    store_act_bf<true>(d0, d1, d2, b1, act, LDA, wv * 16, lane);
    __syncthreads();

    v8f e0 = {}, e1 = {}, e2 = {};
    gemm48<128>(act, LDA, w2T, wv, lane, e0, e1, e2);
    float part = 0.f;
    #pragma unroll
    for (int q = 0; q < 8; ++q) part += e0[q] + e1[q] + e2[q];
    red[wv * 32 + lane] = part;       // partial column sum (24 rows per lane)
    __syncthreads();

    if (t < HD) {
        const int w = t >> 4, l = t & 15;
        const float csum = red[w * 32 + l] + red[w * 32 + 16 + l];
        nodev[t] = node_h[(size_t)i * HD + t] + csum + (float)KNB * b2[t];
    }
    __syncthreads();

    // LN1 over 128
    float x = (t < HD) ? nodev[t] : 0.f;
    red[t] = x; __syncthreads();
    for (int s = 128; s > 0; s >>= 1) { if (t < s) red[t] += red[t + s]; __syncthreads(); }
    const float m1 = red[0] * (1.f / HD);
    __syncthreads();
    const float dx = (t < HD) ? (x - m1) : 0.f;
    red[t] = dx * dx; __syncthreads();
    for (int s = 128; s > 0; s >>= 1) { if (t < s) red[t] += red[t + s]; __syncthreads(); }
    const float rs1 = rsqrtf(red[0] * (1.f / HD) + EPS);
    if (t < HD) {
        const float n1 = dx * rs1 * ln1_g[t] + ln1_b[t];
        node1f[(size_t)i * HD + t] = n1;
        node1b[(size_t)i * HD + t] = (__bf16)n1;
    }
}

// ---------------- phase 2: batched FF (48 nodes/block) + LN2 ----------------
__global__ __launch_bounds__(256) void ff_kernel(
    const __bf16* __restrict__ node1b, const float* __restrict__ node1f,
    const __bf16* __restrict__ f0T, const __bf16* __restrict__ f1T,
    const float* __restrict__ ff_b0, const float* __restrict__ ff_b1,
    const float* __restrict__ ln2_g, const float* __restrict__ ln2_b,
    float* __restrict__ out_node, __bf16* __restrict__ nodefb)
{
    __shared__ __attribute__((aligned(16))) __bf16 Xn[KNB * LDA];
    __shared__ __attribute__((aligned(16))) __bf16 actf[KNB * LDF];
    __shared__ __attribute__((aligned(16))) float yb[KNB * LDD];

    const int i0 = blockIdx.x * KNB;
    const int t = threadIdx.x;
    const int lane = t & 31, wv = t >> 5;

    for (int e = t; e < KNB * HD; e += 256) {
        const int r = e >> 7, c = e & 127;
        Xn[r * LDA + c] = node1b[((size_t)i0 + r) * HD + c];
    }
    __syncthreads();

    // [48,128] @ ff_w0T([512][128]) -> [48,512], gelu
    v8f acc[4][3] = {};
    #pragma unroll
    for (int kt = 0; kt < 128; kt += 32) {
        v16bf a0 = frag_ld(Xn, LDA, 0,  kt, lane);
        v16bf a1 = frag_ld(Xn, LDA, 16, kt, lane);
        v16bf a2 = frag_ld(Xn, LDA, 32, kt, lane);
        #pragma unroll
        for (int s = 0; s < 4; ++s) {
            v16bf b = frag_ld(f0T, 128, s * 128 + wv * 16, kt, lane);
            acc[s][0] = wmma_bf16(a0, b, acc[s][0]);
            acc[s][1] = wmma_bf16(a1, b, acc[s][1]);
            acc[s][2] = wmma_bf16(a2, b, acc[s][2]);
        }
    }
    #pragma unroll
    for (int s = 0; s < 4; ++s)
        store_act_bf<true>(acc[s][0], acc[s][1], acc[s][2], ff_b0, actf, LDF, s * 128 + wv * 16, lane);
    __syncthreads();

    // [48,512] @ ff_w1T([128][512]) -> [48,128]
    v8f c0 = {}, c1 = {}, c2 = {};
    gemm48<512>(actf, LDF, f1T, wv, lane, c0, c1, c2);
    store_f32_ld(c0, c1, c2, ff_b1, yb, LDD, wv * 16, lane);
    __syncthreads();

    // per-node (row) residual + LN2
    for (int r = wv; r < KNB; r += 8) {
        const size_t nb = ((size_t)i0 + r) * HD;
        float xs[4]; float sm = 0.f;
        #pragma unroll
        for (int q = 0; q < 4; ++q) {
            const int c = lane * 4 + q;
            const float xx = node1f[nb + c] + yb[r * LDD + c];
            xs[q] = xx; sm += xx;
        }
        sm = wave_sum(sm);
        const float mean = sm * (1.f / HD);
        float vq = 0.f;
        #pragma unroll
        for (int q = 0; q < 4; ++q) { const float d = xs[q] - mean; vq += d * d; }
        vq = wave_sum(vq);
        const float rs = rsqrtf(vq * (1.f / HD) + EPS);
        #pragma unroll
        for (int q = 0; q < 4; ++q) {
            const int c = lane * 4 + q;
            const float nf = (xs[q] - mean) * rs * ln2_g[c] + ln2_b[c];
            out_node[nb + c] = nf;
            nodefb[nb + c] = (__bf16)nf;
        }
    }
}

// ---------------- phase 3: edge MLP + row LN ----------------
__global__ __launch_bounds__(256) void edge_kernel(
    const float* __restrict__ edge_h, const int* __restrict__ edge_idx,
    const __bf16* __restrict__ nodefb,
    const __bf16* __restrict__ e0T, const __bf16* __restrict__ e1T, const __bf16* __restrict__ e2T,
    const float* __restrict__ b0, const float* __restrict__ b1, const float* __restrict__ b2,
    const float* __restrict__ lne_g, const float* __restrict__ lne_b,
    float* __restrict__ out_edge)
{
    __shared__ __attribute__((aligned(16))) __bf16 X2[KNB * LDX];
    __shared__ __attribute__((aligned(16))) __bf16 act[KNB * LDA];
    __shared__ __attribute__((aligned(16))) float delta[KNB * LDD];

    const int i = blockIdx.x;
    const int t = threadIdx.x;
    const int lane = t & 31, wv = t >> 5;

    for (int e = t; e < KNB * HD; e += 256) {
        const int r = e >> 7, c = e & 127;
        const int j = edge_idx[i * KNB + r];
        X2[r * LDX + c]          = nodefb[(size_t)i * HD + c];
        X2[r * LDX + HD + c]     = nodefb[(size_t)j * HD + c];
        X2[r * LDX + 2 * HD + c] = (__bf16)edge_h[((size_t)i * KNB + r) * HD + c];
    }
    __syncthreads();

    v8f c0 = {}, c1 = {}, c2 = {};
    gemm48<384>(X2, LDX, e0T, wv, lane, c0, c1, c2);
    store_act_bf<true>(c0, c1, c2, b0, act, LDA, wv * 16, lane);
    __syncthreads();

    v8f d0 = {}, d1 = {}, d2 = {};
    gemm48<128>(act, LDA, e1T, wv, lane, d0, d1, d2);
    __syncthreads();
    store_act_bf<true>(d0, d1, d2, b1, act, LDA, wv * 16, lane);
    __syncthreads();

    v8f g0 = {}, g1 = {}, g2 = {};
    gemm48<128>(act, LDA, e2T, wv, lane, g0, g1, g2);
    store_f32_ld(g0, g1, g2, b2, delta, LDD, wv * 16, lane);
    __syncthreads();

    // row-wise LN(edge_h + delta)
    for (int r = wv; r < KNB; r += 8) {
        const size_t eb = ((size_t)i * KNB + r) * HD;
        float xs[4]; float sm = 0.f;
        #pragma unroll
        for (int q = 0; q < 4; ++q) {
            const int c = lane * 4 + q;
            const float xx = edge_h[eb + c] + delta[r * LDD + c];
            xs[q] = xx; sm += xx;
        }
        sm = wave_sum(sm);
        const float mean = sm * (1.f / HD);
        float vq = 0.f;
        #pragma unroll
        for (int q = 0; q < 4; ++q) { const float d = xs[q] - mean; vq += d * d; }
        vq = wave_sum(vq);
        const float rs = rsqrtf(vq * (1.f / HD) + EPS);
        #pragma unroll
        for (int q = 0; q < 4; ++q) {
            const int c = lane * 4 + q;
            out_edge[eb + c] = (xs[q] - mean) * rs * lne_g[c] + lne_b[c];
        }
    }
}

extern "C" void kernel_launch(void* const* d_in, const int* in_sizes, int n_in,
                              void* d_out, int out_size, void* d_ws, size_t ws_size,
                              hipStream_t stream) {
    (void)in_sizes; (void)n_in; (void)out_size; (void)ws_size;
    const float* node_h  = (const float*)d_in[0];
    const float* edge_h  = (const float*)d_in[1];
    const float* msg_w0  = (const float*)d_in[2];
    const float* msg_b0  = (const float*)d_in[3];
    const float* msg_w1  = (const float*)d_in[4];
    const float* msg_b1  = (const float*)d_in[5];
    const float* msg_w2  = (const float*)d_in[6];
    const float* msg_b2  = (const float*)d_in[7];
    const float* ff_w0   = (const float*)d_in[8];
    const float* ff_b0   = (const float*)d_in[9];
    const float* ff_w1   = (const float*)d_in[10];
    const float* ff_b1   = (const float*)d_in[11];
    const float* edge_w0 = (const float*)d_in[12];
    const float* edge_b0 = (const float*)d_in[13];
    const float* edge_w1 = (const float*)d_in[14];
    const float* edge_b1 = (const float*)d_in[15];
    const float* edge_w2 = (const float*)d_in[16];
    const float* edge_b2 = (const float*)d_in[17];
    const float* ln1_g   = (const float*)d_in[18];
    const float* ln1_b   = (const float*)d_in[19];
    const float* ln2_g   = (const float*)d_in[20];
    const float* ln2_b   = (const float*)d_in[21];
    const float* lne_g   = (const float*)d_in[22];
    const float* lne_b   = (const float*)d_in[23];
    const int*   edge_idx= (const int*)d_in[24];

    float* out_node = (float*)d_out;
    float* out_edge = out_node + (size_t)RES * HD;

    char* wsp = (char*)d_ws;
    size_t off = 0;
    auto alloc = [&](size_t bytes) -> void* {
        void* p = wsp + off;
        off += (bytes + 255) & ~((size_t)255);
        return p;
    };
    __bf16* nodeb0 = (__bf16*)alloc((size_t)RES * HD * 2);
    __bf16* node1b = (__bf16*)alloc((size_t)RES * HD * 2);
    __bf16* nodefb = (__bf16*)alloc((size_t)RES * HD * 2);
    float*  node1f = (float*) alloc((size_t)RES * HD * 4);
    __bf16* w0T = (__bf16*)alloc(384 * 128 * 2);
    __bf16* w1T = (__bf16*)alloc(128 * 128 * 2);
    __bf16* w2T = (__bf16*)alloc(128 * 128 * 2);
    __bf16* e0T = (__bf16*)alloc(384 * 128 * 2);
    __bf16* e1T = (__bf16*)alloc(128 * 128 * 2);
    __bf16* e2T = (__bf16*)alloc(128 * 128 * 2);
    __bf16* f0T = (__bf16*)alloc(128 * 512 * 2);
    __bf16* f1T = (__bf16*)alloc(512 * 128 * 2);

    convert_bf16_kernel<<<(RES * HD + 255) / 256, 256, 0, stream>>>(node_h, nodeb0, RES * HD);
    transpose_bf16_kernel<<<(384 * 128 + 255) / 256, 256, 0, stream>>>(msg_w0, w0T, 384, 128);
    transpose_bf16_kernel<<<64, 256, 0, stream>>>(msg_w1, w1T, 128, 128);
    transpose_bf16_kernel<<<64, 256, 0, stream>>>(msg_w2, w2T, 128, 128);
    transpose_bf16_kernel<<<(384 * 128 + 255) / 256, 256, 0, stream>>>(edge_w0, e0T, 384, 128);
    transpose_bf16_kernel<<<64, 256, 0, stream>>>(edge_w1, e1T, 128, 128);
    transpose_bf16_kernel<<<64, 256, 0, stream>>>(edge_w2, e2T, 128, 128);
    transpose_bf16_kernel<<<256, 256, 0, stream>>>(ff_w0, f0T, 128, 512);
    transpose_bf16_kernel<<<256, 256, 0, stream>>>(ff_w1, f1T, 512, 128);

    msg_node_kernel<<<RES, 256, 0, stream>>>(node_h, edge_h, edge_idx, nodeb0,
        w0T, w1T, w2T, msg_b0, msg_b1, msg_b2, ln1_g, ln1_b, node1f, node1b);
    ff_kernel<<<RES / KNB, 256, 0, stream>>>(node1b, node1f, f0T, f1T,
        ff_b0, ff_b1, ln2_g, ln2_b, out_node, nodefb);
    edge_kernel<<<RES, 256, 0, stream>>>(edge_h, edge_idx, nodefb,
        e0T, e1T, e2T, edge_b0, edge_b1, edge_b2, lne_g, lne_b, out_edge);
}
